// AdaptiveGraphDeformation_45990509806147
// MI455X (gfx1250) — matrix operand order repaired and checked by reading, not compiled, not run on hardware
//
#include <hip/hip_runtime.h>
#include <hip/hip_bf16.h>
#include <math.h>

#define B_   16
#define H_   128
#define W_   128
#define C_   192
#define HID  64
#define HP   64
#define WP   64
#define LP   (HP * WP)

typedef __attribute__((ext_vector_type(16))) _Float16 v16h;
typedef __attribute__((ext_vector_type(8)))  _Float16 v8h;
typedef __attribute__((ext_vector_type(8)))  float    v8f;

__device__ __forceinline__ float gelu_exact(float x) {
    return 0.5f * x * (1.0f + erff(x * 0.70710678118654752f));
}
__device__ __forceinline__ float softplus_t(float x) {
    return (x > 20.0f) ? x : log1pf(expf(x));
}

// ---------------------------------------------------------------------------
// Kernel 1: fused  avg-pool(2x2)  ->  GEMM1 (WMMA f16->f32)  -> GELU
//           -> GEMM2 (64x4) -> tanh/softplus -> deform params + coords
// grid: B*HP blocks (one pooled row of 64 positions per block), 128 threads
// wave w handles M-tile rows [w*16, w*16+16)
// ---------------------------------------------------------------------------
__global__ __launch_bounds__(128)
void deform_mlp_kernel(const float* __restrict__ feat,
                       const float* __restrict__ W1,
                       const float* __restrict__ b1,
                       const float* __restrict__ W2,
                       const float* __restrict__ b2,
                       const float* __restrict__ wcoef,
                       float* __restrict__ outParams,   // (B, LP, 4)
                       float* __restrict__ outCoords)   // (B, LP, 2)
{
    __shared__ _Float16 As[64][200];   // pooled tile (f16), row = wp, col = k
    __shared__ _Float16 Bs[64][200];   // W1 transposed (f16), row = n, col = k
    __shared__ float    Hs[4][16][68]; // post-GELU hidden, per wave
    __shared__ float    W2s[64][4];
    __shared__ float    b1s[HID];
    __shared__ float    b2s[4];
    __shared__ float    wcs[4];

    const int tid = threadIdx.x;
    const int b   = blockIdx.x / HP;
    const int hp  = blockIdx.x % HP;

    // ---- stage weights ----
    for (int i = tid; i < C_ * HID; i += 128) {
        int n = i / C_, k = i - n * C_;
        Bs[n][k] = (_Float16)W1[(size_t)k * HID + n];   // transpose: (n, k)
    }
    for (int i = tid; i < 256; i += 128) { W2s[i >> 2][i & 3] = W2[i]; }
    if (tid < HID) b1s[tid] = b1[tid];
    if (tid < 4)   { b2s[tid] = b2[tid]; wcs[tid] = wcoef[tid]; }

    // ---- 2x2 average pool into As (64 pooled positions x 192 ch) ----
    {
        const size_t row0 = ((size_t)(b * H_ + 2 * hp) * W_) * C_;
        const size_t row1 = row0 + (size_t)W_ * C_;
        for (int i = tid; i < 64 * C_; i += 128) {
            int wp = i / C_, c = i - wp * C_;
            size_t o = (size_t)(2 * wp) * C_ + c;
            float s = feat[row0 + o] + feat[row0 + o + C_]
                    + feat[row1 + o] + feat[row1 + o + C_];
            As[wp][c] = (_Float16)(s * 0.25f);
        }
    }
    __syncthreads();

    // ---- GEMM1: (16 x 192) x (192 x 64) per wave via v_wmma_f32_16x16x32_f16
    const int wave  = tid >> 5;
    const int lane  = tid & 31;
    const int lhalf = lane >> 4;     // 0: lanes 0-15, 1: lanes 16-31
    const int lmod  = lane & 15;
    const int arow  = wave * 16 + lmod;

    v8f acc0 = {}, acc1 = {}, acc2 = {}, acc3 = {};
    #pragma unroll
    for (int kk = 0; kk < 6; ++kk) {
        const int kbase = kk * 32;
        // A fragment (contiguous K, matches B): lanes 0-15 -> K 0..15,
        // lanes 16-31 -> K 16..31
        const int aoff = kbase + lhalf * 16;
        v8h alo = *(const v8h*)&As[arow][aoff];
        v8h ahi = *(const v8h*)&As[arow][aoff + 8];
        v16h a;
        #pragma unroll
        for (int i = 0; i < 8; ++i) { a[i] = alo[i]; a[i + 8] = ahi[i]; }
        // B fragments: lane n -> column, K 0..15 (lanes 0-15) / 16..31
        const int bk = kbase + lhalf * 16;
        v16h bf0 = *(const v16h*)&Bs[lmod +  0][bk];
        v16h bf1 = *(const v16h*)&Bs[lmod + 16][bk];
        v16h bf2 = *(const v16h*)&Bs[lmod + 32][bk];
        v16h bf3 = *(const v16h*)&Bs[lmod + 48][bk];
        acc0 = __builtin_amdgcn_wmma_f32_16x16x32_f16(false, a, false, bf0, (short)0, acc0, false, false);
        acc1 = __builtin_amdgcn_wmma_f32_16x16x32_f16(false, a, false, bf1, (short)0, acc1, false, false);
        acc2 = __builtin_amdgcn_wmma_f32_16x16x32_f16(false, a, false, bf2, (short)0, acc2, false, false);
        acc3 = __builtin_amdgcn_wmma_f32_16x16x32_f16(false, a, false, bf3, (short)0, acc3, false, false);
    }

    // ---- bias + GELU, scatter to Hs (D layout: vgpr j -> M=j(+8), lane -> N)
    {
        const int mb = lhalf * 8;
        #pragma unroll
        for (int j = 0; j < 8; ++j) {
            int n0 = lmod;
            Hs[wave][mb + j][n0 +  0] = gelu_exact(acc0[j] + b1s[n0 +  0]);
            Hs[wave][mb + j][n0 + 16] = gelu_exact(acc1[j] + b1s[n0 + 16]);
            Hs[wave][mb + j][n0 + 32] = gelu_exact(acc2[j] + b1s[n0 + 32]);
            Hs[wave][mb + j][n0 + 48] = gelu_exact(acc3[j] + b1s[n0 + 48]);
        }
    }
    // same-wave LDS ops are in-order: safe to read Hs[wave] back without barrier

    // ---- GEMM2 (64 -> 4) + activations + coords: 64 tasks over 32 lanes ----
    #pragma unroll
    for (int t = 0; t < 2; ++t) {
        const int task = lane * 2 + t;
        const int row  = task >> 2;   // 0..15
        const int j    = task & 3;    // output component
        float dp = b2s[j];
        #pragma unroll 8
        for (int n = 0; n < HID; ++n) dp += Hs[wave][row][n] * W2s[n][j];
        dp *= wcs[j];

        const int wp  = wave * 16 + row;
        const int pid = b * LP + hp * WP + wp;
        float val = (j < 2) ? tanhf(dp) : softplus_t(dp);
        outParams[(size_t)pid * 4 + j] = val;

        if (j == 0) {
            float ox = 2.0f * (float)wp / (float)(WP - 1) - 1.0f;
            float dx = ox + val * (2.0f / (float)WP);
            outCoords[(size_t)pid * 2 + 0] = fminf(fmaxf(dx, -1.0f), 1.0f);
        } else if (j == 1) {
            float oy = 2.0f * (float)hp / (float)(HP - 1) - 1.0f;
            float dy = oy + val * (2.0f / (float)HP);
            outCoords[(size_t)pid * 2 + 1] = fminf(fmaxf(dy, -1.0f), 1.0f);
        }
    }
}

// ---------------------------------------------------------------------------
// Kernel 2: bilinear-upsample coords (align_corners) + grid_sample (border)
// grid: B*H blocks (one output row each), 256 threads
// ---------------------------------------------------------------------------
__global__ __launch_bounds__(256)
void resample_kernel(const float* __restrict__ feat,
                     const float* __restrict__ coords,  // (B, HP, WP, 2)
                     float* __restrict__ outFeat)       // (B, H, W, C)
{
    __shared__ int   sx0[W_], sx1[W_], sy0[W_], sy1[W_];
    __shared__ float swx[W_], swy[W_];

    const int tid = threadIdx.x;
    const int b   = blockIdx.x / H_;
    const int y   = blockIdx.x % H_;

    if (tid < W_) {
        const int x = tid;
        // upsample coords map (64x64 -> 128x128), align_corners=True
        const float py = (float)y * (63.0f / 127.0f);
        const float px = (float)x * (63.0f / 127.0f);
        const int jy0 = (int)py, jx0 = (int)px;
        const int jy1 = min(jy0 + 1, HP - 1), jx1 = min(jx0 + 1, WP - 1);
        const float fy = py - (float)jy0, fx = px - (float)jx0;
        const float* cm = coords + (size_t)b * LP * 2;
        float c00x = cm[((size_t)jy0 * WP + jx0) * 2 + 0];
        float c01x = cm[((size_t)jy0 * WP + jx1) * 2 + 0];
        float c10x = cm[((size_t)jy1 * WP + jx0) * 2 + 0];
        float c11x = cm[((size_t)jy1 * WP + jx1) * 2 + 0];
        float c00y = cm[((size_t)jy0 * WP + jx0) * 2 + 1];
        float c01y = cm[((size_t)jy0 * WP + jx1) * 2 + 1];
        float c10y = cm[((size_t)jy1 * WP + jx0) * 2 + 1];
        float c11y = cm[((size_t)jy1 * WP + jx1) * 2 + 1];
        float cx = (c00x + fx * (c01x - c00x)) + fy * ((c10x + fx * (c11x - c10x)) - (c00x + fx * (c01x - c00x)));
        float cy = (c00y + fx * (c01y - c00y)) + fy * ((c10y + fx * (c11y - c10y)) - (c00y + fx * (c01y - c00y)));
        // grid_sample, border padding, align_corners=True
        float ixf = (cx + 1.0f) * 0.5f * (float)(W_ - 1);
        float iyf = (cy + 1.0f) * 0.5f * (float)(H_ - 1);
        ixf = fminf(fmaxf(ixf, 0.0f), (float)(W_ - 1));
        iyf = fminf(fmaxf(iyf, 0.0f), (float)(H_ - 1));
        int ix0 = (int)ixf, iy0 = (int)iyf;
        sx0[x] = ix0;  sx1[x] = min(ix0 + 1, W_ - 1);  swx[x] = ixf - (float)ix0;
        sy0[x] = iy0;  sy1[x] = min(iy0 + 1, H_ - 1);  swy[x] = iyf - (float)iy0;
    }
    __syncthreads();

    const float* fb = feat + (size_t)b * H_ * W_ * C_;
    float*       ob = outFeat + ((size_t)(b * H_ + y)) * W_ * C_;

    #pragma unroll 4
    for (int i = 0; i < (W_ * C_) / 256; ++i) {
        const int e = tid + i * 256;
        const int x = e / C_;
        const int c = e - x * C_;
        const int ix0 = sx0[x], ix1 = sx1[x], iy0 = sy0[x], iy1 = sy1[x];
        const float wx = swx[x], wy = swy[x];
        const float* r0 = fb + ((size_t)iy0 * W_) * C_ + c;
        const float* r1 = fb + ((size_t)iy1 * W_) * C_ + c;
        float v00 = r0[(size_t)ix0 * C_];
        float v01 = r0[(size_t)ix1 * C_];
        float v10 = r1[(size_t)ix0 * C_];
        float v11 = r1[(size_t)ix1 * C_];
        float top = v00 + wx * (v01 - v00);
        float bot = v10 + wx * (v11 - v10);
        ob[e] = top + wy * (bot - top);
    }
}

extern "C" void kernel_launch(void* const* d_in, const int* in_sizes, int n_in,
                              void* d_out, int out_size, void* d_ws, size_t ws_size,
                              hipStream_t stream) {
    const float* feat  = (const float*)d_in[0];
    // d_in[1] = H, d_in[2] = W (python ints; dims are hardcoded)
    const float* W1    = (const float*)d_in[3];
    const float* b1    = (const float*)d_in[4];
    const float* W2    = (const float*)d_in[5];
    const float* b2    = (const float*)d_in[6];
    const float* wcoef = (const float*)d_in[7];

    float* out = (float*)d_out;
    const size_t featN  = (size_t)B_ * H_ * W_ * C_;   // 50331648
    const size_t paramN = (size_t)B_ * LP * 4;         // 262144
    float* outFeat   = out;
    float* outParams = out + featN;
    float* outCoords = out + featN + paramN;

    deform_mlp_kernel<<<B_ * HP, 128, 0, stream>>>(
        feat, W1, b1, W2, b2, wcoef, outParams, outCoords);
    resample_kernel<<<B_ * H_, 256, 0, stream>>>(
        feat, outCoords, outFeat);
}
